// MambaAnglePredictor_40819369181222
// MI455X (gfx1250) — compile-verified, matrix operand order checked by reading, not run on hardware
//
#include <hip/hip_runtime.h>
#include <hip/hip_bf16.h>

// ---------------------------------------------------------------------------
// Mamba2 angle predictor for MI455X (gfx1250, wave32, WMMA everywhere).
//  - Dense GEMMs: v_wmma_f32_16x16x32_f16, 16x64 strip per wave; all fragment
//    loads issued before the WMMA burst so the backend can use partial
//    s_wait_loadcnt; global_prefetch of the next K tile.
//  - Scan: chunked SSD; intra-chunk gram, masked attention, inter-chunk state
//    and local-state update all via WMMA on LDS-staged f16 fragments.
// ---------------------------------------------------------------------------

#define BATCH   4
#define SEQ     4096
#define ROWS    (BATCH * SEQ)        // 16384
#define DMODEL  192
#define DINNER  384
#define DSTATE  64
#define NHEADS  8
#define HEADDIM 48
#define DCONV   4
#define CONVDIM (DINNER + 2 * DSTATE)              // 512
#define DPROJ   (2 * DINNER + 2 * DSTATE + NHEADS) // 904
#define DPROJP  960                                 // padded to 64
#define MIXW    32
#define MIXP    64
#define CL      64                                  // chunk length
#define NC      (SEQ / CL)                          // 64 chunks

typedef __attribute__((ext_vector_type(16))) _Float16 v16h;
typedef __attribute__((ext_vector_type(8)))  float    v8f;

// ---------------------------------------------------------------------------
// WMMA 16-bit A/B fragment layout (ISA 7.12.2): lane L holds row/col (L&15);
// lanes 0-15 carry K {0..7,16..23}, lanes 16-31 carry K {8..15,24..31}.
// ---------------------------------------------------------------------------
__device__ __forceinline__ int koff(int j, int khalf) {
  return (j & 7) + ((j >> 3) << 4) + (khalf << 3);
}

// contiguous-in-K source (row-major, p = &mat[outer*LD + k0])
__device__ __forceinline__ v16h frag_row(const _Float16* __restrict__ p, int khalf) {
  v16h f;
#pragma unroll
  for (int j = 0; j < 16; ++j) f[j] = p[koff(j, khalf)];
  return f;
}

// strided-in-K source (p = &mat[k0*stride + outer], element k at p[k*stride])
__device__ __forceinline__ v16h frag_col(const _Float16* __restrict__ p, int stride,
                                         int khalf) {
  v16h f;
#pragma unroll
  for (int j = 0; j < 16; ++j) f[j] = p[koff(j, khalf) * stride];
  return f;
}

__device__ __forceinline__ v8f wmma_f16(v16h a, v16h b, v8f c) {
  return __builtin_amdgcn_wmma_f32_16x16x32_f16(false, a, false, b, (short)0, c,
                                                false, false);
}

// ---------------------------------------------------------------------------
// Dense GEMM: C[MxN] = act(A[MxK]_f16 * BT[NpadxK]_f16 + bias)
// block = 32 (one wave), each wave computes a 16x64 strip (4 accumulators,
// A fragment reused 4x).  grid = (Npad/64, M/16).  K % 32 == 0, Npad % 64 == 0.
// ---------------------------------------------------------------------------
__global__ void k_gemm_wmma(const _Float16* __restrict__ A,
                            const _Float16* __restrict__ BT,
                            float* __restrict__ Cf,          // optional
                            _Float16* __restrict__ C16,      // optional
                            const float* __restrict__ bias,  // optional
                            int M, int N, int K, int act) {
  const int lane  = threadIdx.x & 31;
  const int khalf = lane >> 4;
  const int l15   = lane & 15;
  const int tm    = blockIdx.y;
  const int tn0   = blockIdx.x * 4;

  const _Float16* ap = A + (size_t)(tm * 16 + l15) * K;
  const _Float16* bp[4];
#pragma unroll
  for (int i = 0; i < 4; ++i)
    bp[i] = BT + (size_t)((tn0 + i) * 16 + l15) * K;

  v8f acc[4] = {};
  for (int k0 = 0; k0 < K; k0 += 32) {
    // Issue all fragment loads first, then the WMMA burst, so the backend can
    // overlap VMEM latency with matrix issue via partial s_wait_loadcnt.
    v16h a  = frag_row(ap + k0, khalf);
    v16h b0 = frag_row(bp[0] + k0, khalf);
    v16h b1 = frag_row(bp[1] + k0, khalf);
    v16h b2 = frag_row(bp[2] + k0, khalf);
    v16h b3 = frag_row(bp[3] + k0, khalf);
    if (k0 + 32 < K) {                       // hint next K tile toward WGP$
      __builtin_prefetch(ap + k0 + 32, 0, 1);
      __builtin_prefetch(bp[0] + k0 + 32, 0, 1);
      __builtin_prefetch(bp[2] + k0 + 32, 0, 1);
    }
    acc[0] = wmma_f16(a, b0, acc[0]);
    acc[1] = wmma_f16(a, b1, acc[1]);
    acc[2] = wmma_f16(a, b2, acc[2]);
    acc[3] = wmma_f16(a, b3, acc[3]);
  }

  const int mbase = tm * 16 + 8 * khalf;
#pragma unroll
  for (int i = 0; i < 4; ++i) {
    int n = (tn0 + i) * 16 + l15;
    if (n < N) {
      float bv = bias ? bias[n] : 0.f;
#pragma unroll
      for (int r = 0; r < 8; ++r) {
        int m = mbase + r;
        float v = acc[i][r] + bv;
        if (act == 1) v = v > 0.f ? v : 0.f;
        if (Cf)  Cf [(size_t)m * N + n] = v;
        if (C16) C16[(size_t)m * N + n] = (_Float16)v;
      }
    }
  }
}

// f32 [K x N] -> f16 transposed [Npad x K], zero-padded rows
__global__ void k_transpose_f16(const float* __restrict__ src,
                                _Float16* __restrict__ dst,
                                int K, int N, int Npad) {
  int i = blockIdx.x * blockDim.x + threadIdx.x;
  if (i >= Npad * K) return;
  int n = i / K, k = i % K;
  dst[i] = (_Float16)((n < N) ? src[(size_t)k * N + n] : 0.f);
}

// h = x @ Wfc + bfc  (K=4 -> VALU), f16 output feeds the in_proj WMMA GEMM
__global__ void k_fc(const float* __restrict__ x, const float* __restrict__ Wfc,
                     const float* __restrict__ bfc, _Float16* __restrict__ h16) {
  int i = blockIdx.x * blockDim.x + threadIdx.x;
  if (i >= ROWS * DMODEL) return;
  int r = i / DMODEL, d = i % DMODEL;
  const float* xr = x + (size_t)r * 4;
  float acc = bfc[d];
#pragma unroll
  for (int c = 0; c < 4; ++c) acc += xr[c] * Wfc[c * DMODEL + d];
  h16[i] = (_Float16)acc;
}

// dt = softplus(raw + dt_bias); dtA = dt * (-exp(A_log))
__global__ void k_dt(const float* __restrict__ zxb, const float* __restrict__ dt_bias,
                     const float* __restrict__ A_log,
                     float* __restrict__ dtb, float* __restrict__ dtAb) {
  int i = blockIdx.x * blockDim.x + threadIdx.x;
  if (i >= ROWS * NHEADS) return;
  int r = i >> 3, hd = i & 7;
  float xv = zxb[(size_t)r * DPROJ + (DINNER + CONVDIM) + hd] + dt_bias[hd];
  float dt = (xv > 20.f) ? xv : log1pf(__expf(xv));
  dtb[i]  = dt;
  dtAb[i] = dt * (-__expf(A_log[hd]));
}

// depthwise causal conv(4) + SiLU
__global__ void k_conv(const float* __restrict__ zxb, const float* __restrict__ conv_w,
                       const float* __restrict__ conv_b, float* __restrict__ xbc) {
  int i = blockIdx.x * blockDim.x + threadIdx.x;
  if (i >= ROWS * CONVDIM) return;
  int r = i / CONVDIM, c = i % CONVDIM;
  int b = r / SEQ, l = r % SEQ;
  float acc = conv_b[c];
#pragma unroll
  for (int w = 0; w < DCONV; ++w) {
    int ls = l - (DCONV - 1) + w;
    if (ls >= 0)
      acc += conv_w[w * CONVDIM + c] *
             zxb[((size_t)(b * SEQ + ls)) * DPROJ + DINNER + c];
  }
  xbc[i] = acc / (1.f + __expf(-acc));
}

// ---------------------------------------------------------------------------
// Chunk-local end state (WMMA): S[48x64] = (w.X)^T[48xCL] @ B[CLx64]
// block 256 = 8 waves; 12 tiles (3 Mp x 4 Nn); K = CL = 64 (2 WMMAs/tile).
// ---------------------------------------------------------------------------
__global__ void k_chunk_state(const float* __restrict__ xbc,
                              const float* __restrict__ dtb,
                              const float* __restrict__ dtAb,
                              float* __restrict__ Sloc, float* __restrict__ Pdec) {
  int chunk = blockIdx.x, bh = blockIdx.y;
  int b = bh >> 3, h = bh & 7;
  int tid = threadIdx.x;
  __shared__ float cum[CL], wls[CL];
  __shared__ _Float16 XwT[HEADDIM][CL];   // A operand: row p, K = t (6 KB)
  __shared__ _Float16 Bh[CL][DSTATE];     // B operand: strided (8 KB)
  int rbase = b * SEQ + chunk * CL;

  if (tid < CL) cum[tid] = dtAb[(size_t)(rbase + tid) * NHEADS + h];
  for (int e = tid; e < CL * DSTATE; e += 256) {
    int t = e >> 6, n = e & 63;
    Bh[t][n] = (_Float16)xbc[(size_t)(rbase + t) * CONVDIM + DINNER + n];
  }
  __syncthreads();
  if (tid == 0) { float s = 0; for (int t = 0; t < CL; ++t) { s += cum[t]; cum[t] = s; } }
  __syncthreads();
  float T = cum[CL - 1];
  if (tid < CL) wls[tid] = __expf(T - cum[tid]) * dtb[(size_t)(rbase + tid) * NHEADS + h];
  __syncthreads();
  for (int e = tid; e < CL * HEADDIM; e += 256) {
    int t = e / HEADDIM, p = e % HEADDIM;
    XwT[p][t] = (_Float16)(wls[t] * xbc[(size_t)(rbase + t) * CONVDIM + h * HEADDIM + p]);
  }
  __syncthreads();

  int wave = tid >> 5, lane = tid & 31, khalf = lane >> 4, l15 = lane & 15;
  float* So = Sloc + ((size_t)bh * NC + chunk) * (HEADDIM * DSTATE);
  for (int tile = wave; tile < 12; tile += 8) {     // wave-uniform
    int tm = tile >> 2, tn = tile & 3;              // tm: p-tiles(3), tn: n-tiles(4)
    v8f acc = {};
#pragma unroll
    for (int k0 = 0; k0 < CL; k0 += 32) {
      v16h a = frag_row(&XwT[tm * 16 + l15][k0], khalf);
      v16h bfr = frag_col(&Bh[k0][tn * 16 + l15], DSTATE, khalf);
      acc = wmma_f16(a, bfr, acc);
    }
    int n = tn * 16 + l15;
#pragma unroll
    for (int r = 0; r < 8; ++r) {
      int p = tm * 16 + r + 8 * khalf;
      So[(size_t)p * DSTATE + n] = acc[r];
    }
  }
  if (tid == 0) Pdec[(size_t)bh * NC + chunk] = __expf(T);
}

// Inter-chunk recurrence (only NC=64 serial steps, all (p,n) parallel)
__global__ void k_state_scan(const float* __restrict__ Sloc,
                             const float* __restrict__ Pdec,
                             float* __restrict__ Sin) {
  int bh = blockIdx.x, tid = threadIdx.x;
  float carry[12];
#pragma unroll
  for (int e = 0; e < 12; ++e) carry[e] = 0.f;
  for (int c = 0; c < NC; ++c) {
    float P = Pdec[(size_t)bh * NC + c];
    const float* Sl = Sloc + ((size_t)bh * NC + c) * (HEADDIM * DSTATE);
    float*       Si = Sin  + ((size_t)bh * NC + c) * (HEADDIM * DSTATE);
#pragma unroll
    for (int e = 0; e < 12; ++e) {
      int idx = tid + 256 * e;
      Si[idx] = carry[e];
      carry[e] = P * carry[e] + Sl[idx];
    }
  }
}

// ---------------------------------------------------------------------------
// Chunk output (WMMA):
//   GEMM1: G[64x64] = C @ B^T              (16 tiles, K=64)
//   mask : Ah[t][s] = (s<=t) ? exp(cum_t-cum_s)*dt_s*G : 0  (f16)
//   concat: Ah[t][64+n] = exp(cum_t)*C[t][n];  B2 = [X ; Sin^T]
//   GEMM2: Y[64x48] = Ah[64x128] @ B2[128x48] (12 tiles, K=128) + D*x skip
// ---------------------------------------------------------------------------
__global__ void k_chunk_out(const float* __restrict__ xbc,
                            const float* __restrict__ dtb,
                            const float* __restrict__ dtAb,
                            const float* __restrict__ Sin,
                            const float* __restrict__ Dparam,
                            float* __restrict__ yb) {
  int chunk = blockIdx.x, bh = blockIdx.y;
  int b = bh >> 3, h = bh & 7;
  int tid = threadIdx.x;
  __shared__ float cum[CL], dts[CL];
  __shared__ _Float16 Ch[CL][DSTATE];              //  8 KB
  __shared__ _Float16 Bh[CL][DSTATE];              //  8 KB
  __shared__ _Float16 Ah[CL][CL + DSTATE];         // 16 KB
  __shared__ _Float16 B2h[CL + DSTATE][HEADDIM];   // 12 KB
  int rbase = b * SEQ + chunk * CL;

  if (tid < CL) {
    cum[tid] = dtAb[(size_t)(rbase + tid) * NHEADS + h];
    dts[tid] = dtb[(size_t)(rbase + tid) * NHEADS + h];
  }
  for (int e = tid; e < CL * DSTATE; e += 256) {
    int t = e >> 6, n = e & 63;
    const float* row = xbc + (size_t)(rbase + t) * CONVDIM + DINNER;
    Bh[t][n] = (_Float16)row[n];
    Ch[t][n] = (_Float16)row[DSTATE + n];
  }
  const float* Si = Sin + ((size_t)bh * NC + chunk) * (HEADDIM * DSTATE);
  for (int e = tid; e < CL * HEADDIM; e += 256) {
    int t = e / HEADDIM, p = e % HEADDIM;
    B2h[t][p] = (_Float16)xbc[(size_t)(rbase + t) * CONVDIM + h * HEADDIM + p];
    // same index pattern reused for Sin^T rows (t plays the role of n here)
    B2h[CL + t][p] = (_Float16)Si[(size_t)p * DSTATE + t];
  }
  __syncthreads();
  if (tid == 0) { float s = 0; for (int t = 0; t < CL; ++t) { s += cum[t]; cum[t] = s; } }
  __syncthreads();

  int wave = tid >> 5, lane = tid & 31, khalf = lane >> 4, l15 = lane & 15;

  // GEMM1 + decay mask -> Ah left half
  for (int tile = wave; tile < 16; tile += 8) {    // wave-uniform
    int tm = tile >> 2, tn = tile & 3;
    v8f acc = {};
#pragma unroll
    for (int k0 = 0; k0 < DSTATE; k0 += 32) {
      v16h a = frag_row(&Ch[tm * 16 + l15][k0], khalf);
      v16h bfr = frag_row(&Bh[tn * 16 + l15][k0], khalf);  // B_s . over n
      acc = wmma_f16(a, bfr, acc);
    }
    int s2 = tn * 16 + l15;
#pragma unroll
    for (int r = 0; r < 8; ++r) {
      int t = tm * 16 + r + 8 * khalf;
      float g = (s2 <= t) ? __expf(cum[t] - cum[s2]) * dts[s2] * acc[r] : 0.f;
      Ah[t][s2] = (_Float16)g;
    }
  }
  // Ah right half: exp(cum_t) * C
  for (int e = tid; e < CL * DSTATE; e += 256) {
    int t = e >> 6, n = e & 63;
    Ah[t][CL + n] = (_Float16)(__expf(cum[t]) * (float)Ch[t][n]);
  }
  __syncthreads();

  // GEMM2: Y = Ah @ B2 (+ D skip)
  float Dp = Dparam[h];
  for (int tile = wave; tile < 12; tile += 8) {    // wave-uniform
    int tm = tile / 3, tn = tile % 3;              // tm: t-tiles(4), tn: p-tiles(3)
    v8f acc = {};
#pragma unroll
    for (int k0 = 0; k0 < CL + DSTATE; k0 += 32) {
      v16h a = frag_row(&Ah[tm * 16 + l15][k0], khalf);
      v16h bfr = frag_col(&B2h[k0][tn * 16 + l15], HEADDIM, khalf);
      acc = wmma_f16(a, bfr, acc);
    }
    int p = tn * 16 + l15;
#pragma unroll
    for (int r = 0; r < 8; ++r) {
      int t = tm * 16 + r + 8 * khalf;
      float v = acc[r] + Dp * (float)B2h[t][p];
      yb[(size_t)(rbase + t) * DINNER + h * HEADDIM + p] = v;
    }
  }
}

// y = rmsnorm(y * silu(z)) * norm_w -> f16
__global__ void k_gatenorm(const float* __restrict__ yb, const float* __restrict__ zxb,
                           const float* __restrict__ norm_w, _Float16* __restrict__ y16) {
  int r = blockIdx.x, tid = threadIdx.x;  // 128 threads x 3
  __shared__ float red[128];
  float vals[3];
  float ss = 0.f;
#pragma unroll
  for (int e = 0; e < 3; ++e) {
    int d = tid + 128 * e;
    float z = zxb[(size_t)r * DPROJ + d];
    float v = yb[(size_t)r * DINNER + d] * (z / (1.f + __expf(-z)));
    vals[e] = v;
    ss += v * v;
  }
  red[tid] = ss;
  __syncthreads();
  for (int s = 64; s > 0; s >>= 1) {
    if (tid < s) red[tid] += red[tid + s];
    __syncthreads();
  }
  float scale = rsqrtf(red[0] / (float)DINNER + 1e-5f);
#pragma unroll
  for (int e = 0; e < 3; ++e) {
    int d = tid + 128 * e;
    y16[(size_t)r * DINNER + d] = (_Float16)(vals[e] * scale * norm_w[d]);
  }
}

// mean over L then dot with Wo
__global__ void k_head(const float* __restrict__ m2, const float* __restrict__ Wo,
                       const float* __restrict__ bo, float* __restrict__ out) {
  int b = blockIdx.x, tid = threadIdx.x;  // 256 = 8 groups x 32 cols
  int col = tid & 31, grp = tid >> 5;
  float acc = 0.f;
  for (int l = grp; l < SEQ; l += 8)
    acc += m2[((size_t)b * SEQ + l) * MIXW + col];
  __shared__ float red[256];
  red[tid] = acc;
  __syncthreads();
  if (grp == 0) {
    float s = 0.f;
    for (int g = 0; g < 8; ++g) s += red[g * 32 + col];
    red[col] = s / (float)SEQ;
  }
  __syncthreads();
  if (tid == 0) {
    float a = bo[0];
    for (int j = 0; j < MIXW; ++j) a += red[j] * Wo[j];
    out[b] = a;
  }
}

// ---------------------------------------------------------------------------
static inline size_t algn(size_t x) { return (x + 255) & ~(size_t)255; }

extern "C" void kernel_launch(void* const* d_in, const int* in_sizes, int n_in,
                              void* d_out, int out_size, void* d_ws, size_t ws_size,
                              hipStream_t stream) {
  const float* x       = (const float*)d_in[0];
  const float* Wfc     = (const float*)d_in[1];
  const float* bfc     = (const float*)d_in[2];
  const float* W_in    = (const float*)d_in[3];
  const float* conv_w  = (const float*)d_in[4];
  const float* conv_b  = (const float*)d_in[5];
  const float* dt_bias = (const float*)d_in[6];
  const float* A_log   = (const float*)d_in[7];
  const float* Dparam  = (const float*)d_in[8];
  const float* norm_w  = (const float*)d_in[9];
  const float* W_out   = (const float*)d_in[10];
  const float* W1      = (const float*)d_in[11];
  const float* b1      = (const float*)d_in[12];
  const float* W2      = (const float*)d_in[13];
  const float* b2      = (const float*)d_in[14];
  const float* Wo      = (const float*)d_in[15];
  const float* bo      = (const float*)d_in[16];
  float* out = (float*)d_out;

  char* w = (char*)d_ws;
  auto carve = [&](size_t bytes) { char* p = w; w += algn(bytes); return p; };

  _Float16*  h16   = (_Float16*) carve((size_t)ROWS * DMODEL * 2);
  _Float16*  WinT  = (_Float16*) carve((size_t)DPROJP * DMODEL * 2);
  _Float16*  WoutT = (_Float16*) carve((size_t)DMODEL * DINNER * 2);
  _Float16*  W1T   = (_Float16*) carve((size_t)MIXP * DMODEL * 2);
  _Float16*  W2T   = (_Float16*) carve((size_t)MIXP * MIXW * 2);
  float*     zxb   = (float*)    carve((size_t)ROWS * DPROJ * 4);
  float*     xbc   = (float*)    carve((size_t)ROWS * CONVDIM * 4);
  float*     dtb   = (float*)    carve((size_t)ROWS * NHEADS * 4);
  float*     dtAb  = (float*)    carve((size_t)ROWS * NHEADS * 4);
  float*     Sloc  = (float*)    carve((size_t)BATCH * NHEADS * NC * HEADDIM * DSTATE * 4);
  float*     Pdec  = (float*)    carve((size_t)BATCH * NHEADS * NC * 4);
  float*     Sin   = (float*)    carve((size_t)BATCH * NHEADS * NC * HEADDIM * DSTATE * 4);
  float*     yb    = (float*)    carve((size_t)ROWS * DINNER * 4);
  _Float16*  y16   = (_Float16*) carve((size_t)ROWS * DINNER * 2);
  _Float16*  h2_16 = (_Float16*) carve((size_t)ROWS * DMODEL * 2);
  _Float16*  m1_16 = (_Float16*) carve((size_t)ROWS * MIXW * 2);
  float*     m2F   = (float*)    carve((size_t)ROWS * MIXW * 4);

  // --- weight prep (transpose + f16, zero-padded to 64-col multiples) ---
  k_transpose_f16<<<(DPROJP * DMODEL + 255) / 256, 256, 0, stream>>>(W_in,  WinT,  DMODEL, DPROJ,  DPROJP);
  k_transpose_f16<<<(DMODEL * DINNER + 255) / 256, 256, 0, stream>>>(W_out, WoutT, DINNER, DMODEL, DMODEL);
  k_transpose_f16<<<(MIXP * DMODEL   + 255) / 256, 256, 0, stream>>>(W1,    W1T,   DMODEL, MIXW,   MIXP);
  k_transpose_f16<<<(MIXP * MIXW     + 255) / 256, 256, 0, stream>>>(W2,    W2T,   MIXW,   MIXW,   MIXP);

  // --- h = x@Wfc + bfc ---
  k_fc<<<(ROWS * DMODEL + 255) / 256, 256, 0, stream>>>(x, Wfc, bfc, h16);

  // --- zxbcdt = h @ W_in (WMMA) ---
  k_gemm_wmma<<<dim3(DPROJP / 64, ROWS / 16), 32, 0, stream>>>(
      h16, WinT, zxb, (_Float16*)nullptr, (const float*)nullptr,
      ROWS, DPROJ, DMODEL, 0);

  // --- dt/softplus, conv+silu ---
  k_dt<<<(ROWS * NHEADS + 255) / 256, 256, 0, stream>>>(zxb, dt_bias, A_log, dtb, dtAb);
  k_conv<<<(ROWS * CONVDIM + 255) / 256, 256, 0, stream>>>(zxb, conv_w, conv_b, xbc);

  // --- chunked SSD scan (WMMA intra-chunk) ---
  k_chunk_state<<<dim3(NC, BATCH * NHEADS), 256, 0, stream>>>(xbc, dtb, dtAb, Sloc, Pdec);
  k_state_scan<<<BATCH * NHEADS, 256, 0, stream>>>(Sloc, Pdec, Sin);
  k_chunk_out<<<dim3(NC, BATCH * NHEADS), 256, 0, stream>>>(xbc, dtb, dtAb, Sin, Dparam, yb);

  // --- gate + RMSNorm ---
  k_gatenorm<<<ROWS, 128, 0, stream>>>(yb, zxb, norm_w, y16);

  // --- h2 = y @ W_out (WMMA) ---
  k_gemm_wmma<<<dim3(DMODEL / 64, ROWS / 16), 32, 0, stream>>>(
      y16, WoutT, (float*)nullptr, h2_16, (const float*)nullptr,
      ROWS, DMODEL, DINNER, 0);

  // --- MLP head (WMMA, bias+ReLU epilogue) ---
  k_gemm_wmma<<<dim3(MIXP / 64, ROWS / 16), 32, 0, stream>>>(
      h2_16, W1T, (float*)nullptr, m1_16, b1, ROWS, MIXW, DMODEL, 1);
  k_gemm_wmma<<<dim3(MIXP / 64, ROWS / 16), 32, 0, stream>>>(
      m1_16, W2T, m2F, (_Float16*)nullptr, b2, ROWS, MIXW, MIXW, 1);

  // --- mean over L, final dot ---
  k_head<<<BATCH, 256, 0, stream>>>(m2F, Wo, bo, out);

  (void)in_sizes; (void)n_in; (void)out_size; (void)ws_size;
}
